// BF15IntLinear_25245817766464
// MI455X (gfx1250) — matrix operand order repaired and checked by reading, not compiled, not run on hardware
//
#include <hip/hip_runtime.h>
#include <hip/hip_bf16.h>
#include <stdint.h>

typedef __attribute__((ext_vector_type(16))) __bf16 v16bf;
typedef __attribute__((ext_vector_type(8)))  float  v8f;

#define M_DIM 512
#define N_DIM 1024
#define K_DIM 1024

// ---------------------------------------------------------------------------
// Pass 1: quantize fp32 -> BF15 (A: mask 0xFFFE0000) and bf16-trunc (W: mask
// 0xFFFF0000). Reference uses floor-of-magnitude, which is exactly a
// sign-magnitude bit truncation, so masking fp32 bits is bit-exact.
// ---------------------------------------------------------------------------
__global__ __launch_bounds__(256)
void bf15_quantize_kernel(const float* __restrict__ x,
                          const float* __restrict__ w,
                          uint16_t* __restrict__ abf,
                          uint16_t* __restrict__ wbf) {
    const int stride = gridDim.x * blockDim.x;
    const int tid = blockIdx.x * blockDim.x + threadIdx.x;
    for (int i = tid; i < M_DIM * K_DIM; i += stride) {
        uint32_t b = __float_as_uint(x[i]) & 0xFFFE0000u;   // BF15 "left"
        abf[i] = (uint16_t)(b >> 16);
    }
    for (int i = tid; i < N_DIM * K_DIM; i += stride) {
        uint32_t b = __float_as_uint(w[i]) & 0xFFFF0000u;   // bf16 trunc
        wbf[i] = (uint16_t)(b >> 16);
    }
}

// ---------------------------------------------------------------------------
// Pass 2: bf16 WMMA GEMM.  Y[M,N] = A[M,K] * W[N,K]^T + bias, out bf16 (RNE).
// Block = 256 threads = 8 waves arranged 2(M) x 4(N); wave tile = 32x32
// (2x2 of 16x16x32 WMMA). Block tile = 64x128. Grid = (N/128, M/64) = (8,8).
// B-columns (K contiguous) are rows of W, so all fragment loads are
// contiguous global_load_b128 — no transpose instructions needed.
// K loop unrolled 4x (16 WMMAs / iter) for load/WMMA overlap; WGP-scope
// prefetch of the next unrolled K-tile on all four row streams.
// ---------------------------------------------------------------------------
__global__ __launch_bounds__(256)
void bf15_wmma_gemm_kernel(const uint16_t* __restrict__ A,   // [M,K] bf15 bits
                           const uint16_t* __restrict__ W,   // [N,K] bf16 bits
                           const float*    __restrict__ bias,
                           uint16_t*       __restrict__ Y) { // [M,N] bf16 bits
    const int lane = threadIdx.x & 31;
    const int wave = threadIdx.x >> 5;
    const int wm   = wave >> 2;        // 0..1
    const int wn   = wave & 3;         // 0..3
    const int half = lane >> 4;        // 0..1
    const int r    = lane & 15;        // 0..15

    const int m0 = blockIdx.y * 64  + wm * 32;
    const int n0 = blockIdx.x * 128 + wn * 32;

    union Frag { uint4 u[2]; v16bf v; };

    v8f acc[2][2];
    #pragma unroll
    for (int mi = 0; mi < 2; ++mi)
        #pragma unroll
        for (int ni = 0; ni < 2; ++ni)
            acc[mi][ni] = (v8f){0.f,0.f,0.f,0.f,0.f,0.f,0.f,0.f};

    const uint16_t* arow0 = A + (size_t)(m0 + r)      * K_DIM;
    const uint16_t* arow1 = A + (size_t)(m0 + 16 + r) * K_DIM;
    const uint16_t* wrow0 = W + (size_t)(n0 + r)      * K_DIM;
    const uint16_t* wrow1 = W + (size_t)(n0 + 16 + r) * K_DIM;

    #pragma unroll 4
    for (int k = 0; k < K_DIM; k += 32) {
        Frag a[2], b[2];
        // A 16x32 bf16 layout: VGPR0-3 <- K = half*8 + [0..7],
        //                      VGPR4-7 <- K = 16 + half*8 + [0..7]
        {
            const uint16_t* p0 = arow0 + k + half * 8;
            const uint16_t* p1 = arow1 + k + half * 8;
            a[0].u[0] = *(const uint4*)(p0);
            a[0].u[1] = *(const uint4*)(p0 + 16);
            a[1].u[0] = *(const uint4*)(p1);
            a[1].u[1] = *(const uint4*)(p1 + 16);
        }
        // B 32x16 bf16 layout: VGPR0-3 <- K = half*8 + [0..7],
        //                      VGPR4-7 <- K = 16 + half*8 + [0..7]
        {
            const uint16_t* p0 = wrow0 + k + half * 8;
            const uint16_t* p1 = wrow1 + k + half * 8;
            b[0].u[0] = *(const uint4*)(p0);
            b[0].u[1] = *(const uint4*)(p0 + 16);
            b[1].u[0] = *(const uint4*)(p1);
            b[1].u[1] = *(const uint4*)(p1 + 16);
        }
        // Prefetch one unrolled iteration (128 elements = 1 cacheline-aligned
        // 256B run per row) ahead, WGP scope (locality=3 -> near-scope fill).
        if ((k & 127) == 0 && k + 128 < K_DIM) {
            __builtin_prefetch(arow0 + k + 128, 0, 3);
            __builtin_prefetch(arow1 + k + 128, 0, 3);
            __builtin_prefetch(wrow0 + k + 128, 0, 3);
            __builtin_prefetch(wrow1 + k + 128, 0, 3);
        }
        #pragma unroll
        for (int mi = 0; mi < 2; ++mi)
            #pragma unroll
            for (int ni = 0; ni < 2; ++ni)
                acc[mi][ni] = __builtin_amdgcn_wmma_f32_16x16x32_bf16(
                    false, a[mi].v, false, b[ni].v,
                    (short)0, acc[mi][ni], false, false);
    }

    // Epilogue: +bias, fp32 -> bf16 round-to-nearest-even.
    // C/D layout: VGPR i, lane l -> (m = (l>>4)*8 + i, n = l&15).
    #pragma unroll
    for (int mi = 0; mi < 2; ++mi) {
        #pragma unroll
        for (int ni = 0; ni < 2; ++ni) {
            const int n = n0 + ni * 16 + r;
            const float bs = bias[n];
            #pragma unroll
            for (int i = 0; i < 8; ++i) {
                const int m = m0 + mi * 16 + half * 8 + i;
                float v = acc[mi][ni][i] + bs;
                uint32_t u  = __float_as_uint(v);
                uint32_t rb = (u + 0x7FFFu + ((u >> 16) & 1u)) >> 16;
                Y[(size_t)m * N_DIM + n] = (uint16_t)rb;
            }
        }
    }
}

extern "C" void kernel_launch(void* const* d_in, const int* in_sizes, int n_in,
                              void* d_out, int out_size, void* d_ws, size_t ws_size,
                              hipStream_t stream) {
    const float* x    = (const float*)d_in[0];   // [4,128,1024] fp32
    const float* w    = (const float*)d_in[1];   // [1024,1024]  fp32
    const float* bias = (const float*)d_in[2];   // [1024]       fp32
    uint16_t* y = (uint16_t*)d_out;              // [512,1024]   bf16 bits

    // Workspace: bf16-quantized A (1 MB) then W (2 MB).
    uint16_t* abf = (uint16_t*)d_ws;
    uint16_t* wbf = abf + (size_t)M_DIM * K_DIM;

    bf15_quantize_kernel<<<dim3(1024), dim3(256), 0, stream>>>(x, w, abf, wbf);

    dim3 grid(N_DIM / 128, M_DIM / 64);          // (8, 8) = 64 workgroups
    bf15_wmma_gemm_kernel<<<grid, dim3(256), 0, stream>>>(abf, wbf, bias, y);
}
